// CrossAttention_vv_32272384262209
// MI455X (gfx1250) — matrix operand order repaired and checked by reading, not compiled, not run on hardware
//
#include <hip/hip_runtime.h>
#include <hip/hip_bf16.h>

typedef __attribute__((ext_vector_type(16))) _Float16 v16h;
typedef __attribute__((ext_vector_type(8)))  _Float16 v8h;
typedef __attribute__((ext_vector_type(8)))  float    v8f;

#define NB  8      // batch
#define NC  256    // channels
#define NS  1024   // sequence (32*32)
#define NH_ 8      // heads
#define NDK 64     // head dim
#define ND  512    // NH*DK

// ---------------- WMMA fragment helpers (layouts per CDNA5 ISA 7.12.2) ----

// A: 16x32 f16.  lane l holds row m=l&15; VGPR v: K = (v>>2)*16 + (l>>4)*8 + (v&3)*2 + j
// => elements 0..7  = K kb..kb+7   (kb=(l>>4)*8),  elements 8..15 = K kb+16..kb+23
__device__ __forceinline__ v16h load_a_frag(const _Float16* src, int ld) {
    int l  = threadIdx.x & 31;
    int r  = l & 15;
    int kb = (l >> 4) << 3;
    const _Float16* p = src + r * ld + kb;
    v8h lo = *(const v8h*)(p);
    v8h hi = *(const v8h*)(p + 16);
    v16h a;
#pragma unroll
    for (int i = 0; i < 8; ++i) { a[i] = lo[i]; a[i + 8] = hi[i]; }
    return a;
}

// B: 32x16 f16, src stored column-major [N][K] with leading dim ld.
// lane l holds col n=l&15; elements 0..7 = K kb..kb+7 (kb=(l>>4)*8), elements 8..15 = K kb+16..kb+23.
__device__ __forceinline__ v16h load_b_frag(const _Float16* src, int ld) {
    int l  = threadIdx.x & 31;
    int n  = l & 15;
    int kb = (l >> 4) << 3;
    const _Float16* p = src + n * ld + kb;
    v8h lo = *(const v8h*)(p);
    v8h hi = *(const v8h*)(p + 16);
    v16h b;
#pragma unroll
    for (int i = 0; i < 8; ++i) { b[i] = lo[i]; b[i + 8] = hi[i]; }
    return b;
}

__device__ __forceinline__ v8f wmma16(v16h a, v16h b, v8f c) {
    return __builtin_amdgcn_wmma_f32_16x16x32_f16(false, a, false, b,
                                                  (short)0, c, false, false);
}

// ---------------- Kernel 0a: input convert+transpose [C,S]f32 -> [S,C]f16 --

__global__ void k_convert_x(const float* __restrict__ q, const float* __restrict__ k,
                            const float* __restrict__ v1, const float* __restrict__ v2,
                            _Float16* __restrict__ xh) {
    int tb = blockIdx.y;            // tensor*8 + b
    int tensor = tb >> 3, b = tb & 7;
    const float* src = (tensor == 0) ? q : (tensor == 1) ? k : (tensor == 2) ? v1 : v2;
    src += (size_t)b * NC * NS;
    _Float16* dst = xh + (size_t)tb * NS * NC;

    int s0 = (blockIdx.x & 31) * 32;
    int c0 = (blockIdx.x >> 5) * 32;
    __shared__ float tile[32][33];
    int j = threadIdx.x & 31, i0 = threadIdx.x >> 5;
#pragma unroll
    for (int r = 0; r < 4; ++r) {
        int i = i0 + r * 8;
        tile[i][j] = src[(size_t)(c0 + i) * NS + s0 + j];
    }
    __syncthreads();
#pragma unroll
    for (int r = 0; r < 4; ++r) {
        int i = i0 + r * 8;
        dst[(size_t)(s0 + i) * NC + c0 + j] = (_Float16)tile[j][i];
    }
}

// ---------------- Kernel 0b: weight convert+transpose [R,Cc]f32 -> [Cc,R]f16

__global__ void k_convert_w(const float* __restrict__ src, int R, int Cc,
                            _Float16* __restrict__ dst) {
    int idx = blockIdx.x * blockDim.x + threadIdx.x;
    if (idx < R * Cc) {
        int i = idx / Cc, j = idx % Cc;
        dst[(size_t)j * R + i] = (_Float16)src[idx];
    }
}

// ---------------- Kernel 1: projections X^T @ W, 32x32 per wave (2x2 tiles)
// Q,K -> [b,h,s,dk] f16 ;  V1,V2 -> [b,h,dk,s] f16 (transposed for attn B-op)

__global__ __launch_bounds__(128)
void k_proj(const _Float16* __restrict__ Xh,   // 4 x [B][S][C]
            const _Float16* __restrict__ Wt,   // 4 x [D][C] (col-major B op)
            _Float16* __restrict__ Qb, _Float16* __restrict__ Kb,
            _Float16* __restrict__ V1t, _Float16* __restrict__ V2t) {
    int w  = blockIdx.x * 4 + (threadIdx.x >> 5);
    int nt = w & 15;            // 16 tiles of 32 over D=512
    int mt = (w >> 4) & 31;     // 32 tiles of 32 over S=1024
    int b  = (w >> 9) & 7;
    int p  = (w >> 12) & 3;

    const _Float16* A0 = Xh + ((size_t)(p * NB + b) * NS + mt * 32) * NC;
    const _Float16* A1 = A0 + 16 * NC;
    const _Float16* B0 = Wt + (size_t)p * ND * NC + (size_t)(nt * 32) * NC;
    const _Float16* B1 = B0 + 16 * NC;

    v8f acc[2][2];
#pragma unroll
    for (int i = 0; i < 2; ++i)
#pragma unroll
        for (int j = 0; j < 2; ++j)
            acc[i][j] = (v8f){0.f,0.f,0.f,0.f,0.f,0.f,0.f,0.f};

#pragma unroll
    for (int kc = 0; kc < NC / 32; ++kc) {
        v16h a0 = load_a_frag(A0 + kc * 32, NC);
        v16h a1 = load_a_frag(A1 + kc * 32, NC);
        v16h b0 = load_b_frag(B0 + kc * 32, NC);
        v16h b1 = load_b_frag(B1 + kc * 32, NC);
        acc[0][0] = wmma16(a0, b0, acc[0][0]);
        acc[0][1] = wmma16(a0, b1, acc[0][1]);
        acc[1][0] = wmma16(a1, b0, acc[1][0]);
        acc[1][1] = wmma16(a1, b1, acc[1][1]);
    }

    int l = threadIdx.x & 31;
    int n = l & 15, mb = (l >> 4) * 8;
    int h = nt >> 1, dk0 = (nt & 1) * 32;   // 32 cols stay within one head

    if (p <= 1) {
        _Float16* dst = (p == 0) ? Qb : Kb;
        size_t base = (size_t)(b * NH_ + h) * NS * NDK;
#pragma unroll
        for (int sm = 0; sm < 2; ++sm)
#pragma unroll
            for (int sn = 0; sn < 2; ++sn)
#pragma unroll
                for (int v = 0; v < 8; ++v) {
                    int s = mt * 32 + sm * 16 + mb + v;
                    dst[base + (size_t)s * NDK + dk0 + sn * 16 + n] =
                        (_Float16)acc[sm][sn][v];
                }
    } else {
        _Float16* dst = (p == 2) ? V1t : V2t;
#pragma unroll
        for (int sn = 0; sn < 2; ++sn) {
            int dk = dk0 + sn * 16 + n;
#pragma unroll
            for (int sm = 0; sm < 2; ++sm) {
                size_t base = ((size_t)(b * NH_ + h) * NDK + dk) * NS
                            + mt * 32 + sm * 16 + mb;
                v8h pk;
#pragma unroll
                for (int v = 0; v < 8; ++v) pk[v] = (_Float16)acc[sm][sn][v];
                *(v8h*)(dst + base) = pk;   // 16B aligned
            }
        }
    }
}

// ---------------- Kernel 2: shared-softmax attention, two value streams ----
// One wave = 32 query rows of one (b,h): K/V fragments reused across both
// 16-row groups (24 WMMAs per 32-key chunk on shared operands).

__global__ __launch_bounds__(128)
void k_attn(const _Float16* __restrict__ Qb, const _Float16* __restrict__ Kb,
            const _Float16* __restrict__ V1t, const _Float16* __restrict__ V2t,
            _Float16* __restrict__ AO1, _Float16* __restrict__ AO2) {
    int widx = threadIdx.x >> 5;
    int w    = blockIdx.x * 4 + widx;
    int q32  = w & 31;          // 32-query tile
    int bh   = w >> 5;          // 0..63
    int b = bh >> 3, h = bh & 7;

    const _Float16* Qp  = Qb  + ((size_t)bh * NS + q32 * 32) * NDK;
    const _Float16* Kp  = Kb  + (size_t)bh * NS * NDK;
    const _Float16* V1p = V1t + (size_t)bh * NDK * NS;
    const _Float16* V2p = V2t + (size_t)bh * NDK * NS;

    __shared__ __align__(16) _Float16 Plds[4][2 * 16 * 32];
    _Float16* myP = &Plds[widx][0];

    v16h qa[2][2];
#pragma unroll
    for (int u = 0; u < 2; ++u) {
        qa[u][0] = load_a_frag(Qp + u * 16 * NDK, NDK);
        qa[u][1] = load_a_frag(Qp + u * 16 * NDK + 32, NDK);
    }

    v8f o1[2][4], o2[2][4];
#pragma unroll
    for (int u = 0; u < 2; ++u)
#pragma unroll
        for (int t = 0; t < 4; ++t) {
            o1[u][t] = (v8f){0.f,0.f,0.f,0.f,0.f,0.f,0.f,0.f};
            o2[u][t] = (v8f){0.f,0.f,0.f,0.f,0.f,0.f,0.f,0.f};
        }
    float m_i[2][8], l_i[2][8];
#pragma unroll
    for (int u = 0; u < 2; ++u)
#pragma unroll
        for (int v = 0; v < 8; ++v) { m_i[u][v] = -1e30f; l_i[u][v] = 0.f; }

    int l = threadIdx.x & 31;
    int n = l & 15, mb = (l >> 4) * 8;
    const v8f z = {0.f,0.f,0.f,0.f,0.f,0.f,0.f,0.f};

    for (int jb = 0; jb < NS; jb += 32) {
        if (jb + 32 < NS) __builtin_prefetch(Kp + (size_t)(jb + 32) * NDK, 0, 1);

        // K fragments for keys jb..jb+31, shared by both query row-groups
        v16h kA = load_b_frag(Kp + (size_t)jb * NDK, NDK);
        v16h kB = load_b_frag(Kp + (size_t)jb * NDK + 32, NDK);
        v16h kC = load_b_frag(Kp + (size_t)(jb + 16) * NDK, NDK);
        v16h kD = load_b_frag(Kp + (size_t)(jb + 16) * NDK + 32, NDK);

#pragma unroll
        for (int u = 0; u < 2; ++u) {
            v8f s0 = wmma16(qa[u][0], kA, z);
            s0     = wmma16(qa[u][1], kB, s0);
            v8f s1 = wmma16(qa[u][0], kC, z);
            s1     = wmma16(qa[u][1], kD, s1);

            // online softmax (C-frag row = mb+v, spread across 16-lane halves)
#pragma unroll
            for (int v = 0; v < 8; ++v) {
                float a0 = s0[v] * 0.125f;      // 1/sqrt(64)
                float a1 = s1[v] * 0.125f;
                float cm = fmaxf(a0, a1);
                cm = fmaxf(cm, __shfl_xor(cm, 1, 32));
                cm = fmaxf(cm, __shfl_xor(cm, 2, 32));
                cm = fmaxf(cm, __shfl_xor(cm, 4, 32));
                cm = fmaxf(cm, __shfl_xor(cm, 8, 32));
                float nm = fmaxf(m_i[u][v], cm);
                float sc = __expf(m_i[u][v] - nm);
                float p0 = __expf(a0 - nm);
                float p1 = __expf(a1 - nm);
                float rs = p0 + p1;
                rs += __shfl_xor(rs, 1, 32);
                rs += __shfl_xor(rs, 2, 32);
                rs += __shfl_xor(rs, 4, 32);
                rs += __shfl_xor(rs, 8, 32);
                l_i[u][v] = l_i[u][v] * sc + rs;
                m_i[u][v] = nm;
#pragma unroll
                for (int t = 0; t < 4; ++t) { o1[u][t][v] *= sc; o2[u][t][v] *= sc; }
                myP[u * 512 + (mb + v) * 32 + n]      = (_Float16)p0;
                myP[u * 512 + (mb + v) * 32 + n + 16] = (_Float16)p1;
            }
        }
        __syncthreads();   // uniform trip count across all 4 waves

        // O += P(16x32) @ V(32x64): V fragments shared by both row-groups
        v16h pa0 = load_a_frag(myP, 32);
        v16h pa1 = load_a_frag(myP + 512, 32);
#pragma unroll
        for (int t = 0; t < 4; ++t) {
            v16h vb1 = load_b_frag(V1p + (size_t)(t * 16) * NS + jb, NS);
            o1[0][t] = wmma16(pa0, vb1, o1[0][t]);
            o1[1][t] = wmma16(pa1, vb1, o1[1][t]);
            v16h vb2 = load_b_frag(V2p + (size_t)(t * 16) * NS + jb, NS);
            o2[0][t] = wmma16(pa0, vb2, o2[0][t]);
            o2[1][t] = wmma16(pa1, vb2, o2[1][t]);
        }
        __syncthreads();
    }

    // normalize and emit [b,s,d] f16 (d = h*64 + t*16 + n)
#pragma unroll
    for (int u = 0; u < 2; ++u)
#pragma unroll
        for (int v = 0; v < 8; ++v) {
            float inv = 1.0f / l_i[u][v];
            int s = q32 * 32 + u * 16 + mb + v;
            size_t row = ((size_t)b * NS + s) * ND + h * NDK;
#pragma unroll
            for (int t = 0; t < 4; ++t) {
                AO1[row + t * 16 + n] = (_Float16)(o1[u][t][v] * inv);
                AO2[row + t * 16 + n] = (_Float16)(o2[u][t][v] * inv);
            }
        }
}

// ---------------- Kernel 3: output projection AO @ Wo + bo, 32x32 per wave -

__global__ __launch_bounds__(128)
void k_outproj(const _Float16* __restrict__ AO1, const _Float16* __restrict__ AO2,
               const _Float16* __restrict__ Woc,  // [C][D] col-major B op
               const float* __restrict__ bo, float* __restrict__ out) {
    int w  = blockIdx.x * 4 + (threadIdx.x >> 5);
    int nt = w & 7;             // 8 tiles of 32 over C=256
    int mt = (w >> 3) & 31;     // 32 tiles of 32 over S=1024
    int b  = (w >> 8) & 7;
    int which = (w >> 11) & 1;

    const _Float16* AO = which ? AO2 : AO1;
    const _Float16* A0 = AO + ((size_t)b * NS + mt * 32) * ND;
    const _Float16* A1 = A0 + 16 * ND;
    const _Float16* B0 = Woc + (size_t)(nt * 32) * ND;
    const _Float16* B1 = B0 + 16 * ND;

    v8f acc[2][2];
#pragma unroll
    for (int i = 0; i < 2; ++i)
#pragma unroll
        for (int j = 0; j < 2; ++j)
            acc[i][j] = (v8f){0.f,0.f,0.f,0.f,0.f,0.f,0.f,0.f};

#pragma unroll
    for (int kc = 0; kc < ND / 32; ++kc) {
        v16h a0 = load_a_frag(A0 + kc * 32, ND);
        v16h a1 = load_a_frag(A1 + kc * 32, ND);
        v16h b0 = load_b_frag(B0 + kc * 32, ND);
        v16h b1 = load_b_frag(B1 + kc * 32, ND);
        acc[0][0] = wmma16(a0, b0, acc[0][0]);
        acc[0][1] = wmma16(a0, b1, acc[0][1]);
        acc[1][0] = wmma16(a1, b0, acc[1][0]);
        acc[1][1] = wmma16(a1, b1, acc[1][1]);
    }

    int l = threadIdx.x & 31;
    int n = l & 15, mb = (l >> 4) * 8;
    float* dst = out + (size_t)which * NB * NS * NC;
#pragma unroll
    for (int sn = 0; sn < 2; ++sn) {
        float bias = bo[nt * 32 + sn * 16 + n];
#pragma unroll
        for (int sm = 0; sm < 2; ++sm)
#pragma unroll
            for (int v = 0; v < 8; ++v) {
                int s = mt * 32 + sm * 16 + mb + v;
                dst[((size_t)b * NS + s) * NC + nt * 32 + sn * 16 + n] =
                    acc[sm][sn][v] + bias;
            }
    }
}

// ---------------- Launch ---------------------------------------------------

extern "C" void kernel_launch(void* const* d_in, const int* in_sizes, int n_in,
                              void* d_out, int out_size, void* d_ws, size_t ws_size,
                              hipStream_t stream) {
    const float* q   = (const float*)d_in[0];
    const float* k   = (const float*)d_in[1];
    const float* v1  = (const float*)d_in[2];
    const float* v2  = (const float*)d_in[3];
    const float* Wq  = (const float*)d_in[4];
    const float* Wk  = (const float*)d_in[5];
    const float* Wv1 = (const float*)d_in[6];
    const float* Wv2 = (const float*)d_in[7];
    const float* Wo  = (const float*)d_in[8];
    const float* bo  = (const float*)d_in[9];

    char* ws = (char*)d_ws;
    const size_t XH  = 0;                                  // 4*B*S*C f16 = 16 MB
    const size_t WT  = XH  + (size_t)4 * NB * NS * NC * 2; // 4*D*C f16   = 1 MB
    const size_t WOC = WT  + (size_t)4 * ND * NC * 2;      // C*D f16
    const size_t QB  = WOC + (size_t)NC * ND * 2;          // B*NH*S*DK f16 = 8 MB
    const size_t KB  = QB  + (size_t)NB * NH_ * NS * NDK * 2;
    const size_t V1T = KB  + (size_t)NB * NH_ * NS * NDK * 2;
    const size_t V2T = V1T + (size_t)NB * NH_ * NS * NDK * 2;
    const size_t AO1 = V2T + (size_t)NB * NH_ * NS * NDK * 2;
    const size_t AO2 = AO1 + (size_t)NB * NS * ND * 2;

    _Float16* xh  = (_Float16*)(ws + XH);
    _Float16* wt  = (_Float16*)(ws + WT);
    _Float16* woc = (_Float16*)(ws + WOC);
    _Float16* qb  = (_Float16*)(ws + QB);
    _Float16* kb  = (_Float16*)(ws + KB);
    _Float16* v1t = (_Float16*)(ws + V1T);
    _Float16* v2t = (_Float16*)(ws + V2T);
    _Float16* ao1 = (_Float16*)(ws + AO1);
    _Float16* ao2 = (_Float16*)(ws + AO2);

    // 0a: inputs -> [S,C] f16 (32 s-tiles * 8 c-tiles = 256 blocks, 32 tensors)
    k_convert_x<<<dim3(256, 32), 256, 0, stream>>>(q, k, v1, v2, xh);

    // 0b: weights -> col-major f16
    const int WNE = NC * ND;  // 131072
    k_convert_w<<<(WNE + 255) / 256, 256, 0, stream>>>(Wq,  NC, ND, wt + 0 * (size_t)WNE);
    k_convert_w<<<(WNE + 255) / 256, 256, 0, stream>>>(Wk,  NC, ND, wt + 1 * (size_t)WNE);
    k_convert_w<<<(WNE + 255) / 256, 256, 0, stream>>>(Wv1, NC, ND, wt + 2 * (size_t)WNE);
    k_convert_w<<<(WNE + 255) / 256, 256, 0, stream>>>(Wv2, NC, ND, wt + 3 * (size_t)WNE);
    k_convert_w<<<(WNE + 255) / 256, 256, 0, stream>>>(Wo,  ND, NC, woc);

    // 1: projections (4 proj * 8 b * 32 m-tiles * 16 n-tiles = 16384 waves)
    k_proj<<<4096, 128, 0, stream>>>(xh, wt, qb, kb, v1t, v2t);

    // 2: attention (8 b * 8 h * 32 q-tiles = 2048 waves)
    k_attn<<<512, 128, 0, stream>>>(qb, kb, v1t, v2t, ao1, ao2);

    // 3: output projection (2 * 8 * 32 * 8 = 4096 waves)
    k_outproj<<<1024, 128, 0, stream>>>(ao1, ao2, woc, bo, (float*)d_out);
}
